// DeepReservoir_27049704030566
// MI455X (gfx1250) — matrix operand order repaired and checked by reading, not compiled, not run on hardware
//
#include <hip/hip_runtime.h>

// ---------------------------------------------------------------------------
// MI455X (gfx1250) fused DeepReservoir forward, v2.
//
// Roofline: ~120 GFLOP dense, ~448 MB min HBM -> ~19us floor.
// v2 change vs v1: each wave owns 4 n-tiles and computes 4 stacked m-tiles
// (64 rows/block) per B-fragment load, so every weight fragment is read from
// L2 exactly once per block: weight L2 traffic drops 14 GB -> 3.6 GB.
// Precision: x-side GEMMs use 3-product split-f16 (x hi/lo x W hi/lo);
// h-side GEMMs use 2-product (h_hi x W hi/lo) - h's f16 rounding (~5e-4)
// dominates, keeping ~1e-3 accuracy while fitting 64 rows in 192 KB LDS.
// ---------------------------------------------------------------------------

typedef __attribute__((ext_vector_type(16))) _Float16 v16h;
typedef __attribute__((ext_vector_type(8)))  float    v8f;
typedef __attribute__((ext_vector_type(4)))  _Float16 v4h;

#define BATCH   65536
#define IN_DIM  256
#define UNITS   512
#define MT      64      // rows per block (4 m-tiles, all owned by every wave)
#define TPB     256     // 8 wave32, each owns 4 of the 32 n-tiles

// Workspace layout (element offsets, _Float16):
#define W0HI 0
#define W0LO 131072
#define W1HI 262144
#define W1LO 524288
#define W2HI 786432
#define W2LO 1048576
#define S1HI 1310720
#define S1LO 1441792
#define S2HI 1572864
#define S2LO 1703936
// total 1835008 halfs = 3.5 MB of d_ws

// ---------------------------------------------------------------------------
// Pack kernel: out[b_swizzle(k,n)] = split_f16(W[k,n] * M[k,n]).
// B-fragment order (16x16x32 f16 WMMA): tile (kt,nt) is 512 contiguous halfs;
// lane = (n&15) + ((k&31)>>4)*16 holds K=kt*32+{0..15 | 16..31} at j = k&15.
// ---------------------------------------------------------------------------
__global__ __launch_bounds__(256) void pack_weights(
    const float* __restrict__ W, const float* __restrict__ M,
    _Float16* __restrict__ hi, _Float16* __restrict__ lo, int K, int N)
{
  int t = blockIdx.x * blockDim.x + threadIdx.x;
  if (t >= K * N) return;
  int k = t / N, n = t % N;
  float v = W[t] * M[t];
  _Float16 h = (_Float16)v;
  _Float16 l = (_Float16)(v - (float)h);
  int kk  = k & 31;
  int off = ((k >> 5) * (N >> 4) + (n >> 4)) * 512
          + (((n & 15) + ((kk >> 4) << 4)) << 4)
          + (kk & 15);
  hi[off] = h;
  lo[off] = l;
}

// ---------------------------------------------------------------------------
// A-fragment swizzled offset inside an LDS plane of MT x KDIM halfs.
// 16-bit A 16x32 layout (matches B packing): lane = (m&15) + ((k&31)>>4)*16,
// half j = k&15, tile (mt,kt) = 512 contiguous halfs.
// ---------------------------------------------------------------------------
__device__ __forceinline__ int a_off(int m, int k, int KT /* KDIM/32 */) {
  int kk = k & 31;
  return (((m >> 4) * KT + (k >> 5)) << 9)
       + (((m & 15) + ((kk >> 4) << 4)) << 4)
       + (kk & 15);
}

__device__ __forceinline__ float sine_act(float z, float f, float a, float d) {
  return a * __sinf(f * z) * __expf(-d * __builtin_fabsf(z));
}

// ---------------------------------------------------------------------------
// Multi-m-tile GEMM: acc[m] (m=0..3) += A-tiles x B(hi+lo).
// Ahi/Alo in LDS (A-fragment swizzled); B planes in global (L2-resident).
// split3: 3-product (Ahi Bhi + Ahi Blo + Alo Bhi); else 2-product (Ahi only).
// B fragments are loaded ONCE per kt and reused by all 4 m-tiles.
// ---------------------------------------------------------------------------
__device__ __forceinline__ void gemm_4m(
    v8f acc[4],
    const _Float16* __restrict__ Ahi, const _Float16* __restrict__ Alo,
    int KT, const _Float16* __restrict__ Bhi, const _Float16* __restrict__ Blo,
    int ntile, int NT, int lane, bool split3)
{
  #pragma unroll 4
  for (int kt = 0; kt < KT; ++kt) {
    const int bo = ((kt * NT + ntile) << 9) + (lane << 4);
    v16h bhi = *reinterpret_cast<const v16h*>(Bhi + bo);
    v16h blo = *reinterpret_cast<const v16h*>(Blo + bo);
    #pragma unroll
    for (int m = 0; m < 4; ++m) {
      const int ao = ((m * KT + kt) << 9) + (lane << 4);
      v16h ahi = *reinterpret_cast<const v16h*>(Ahi + ao);
      acc[m] = __builtin_amdgcn_wmma_f32_16x16x32_f16(false, ahi, false, bhi, (short)0, acc[m], false, false);
      acc[m] = __builtin_amdgcn_wmma_f32_16x16x32_f16(false, ahi, false, blo, (short)0, acc[m], false, false);
      if (split3) {
        v16h alo = *reinterpret_cast<const v16h*>(Alo + ao);
        acc[m] = __builtin_amdgcn_wmma_f32_16x16x32_f16(false, alo, false, bhi, (short)0, acc[m], false, false);
      }
    }
  }
}

// ---------------------------------------------------------------------------
// Fused forward. Block = 64 rows x 512 cols per layer.
// Wave w owns n-tiles [4w, 4w+4) and all 4 m-tiles.
// LDS: x hi/lo 64KB + h0_hi 64KB + h1_hi 64KB = 192KB.
// ---------------------------------------------------------------------------
__global__ __launch_bounds__(TPB) void reservoir_fused(
    const float* __restrict__ x, const _Float16* __restrict__ ws,
    const float* __restrict__ b0, const float* __restrict__ b1, const float* __restrict__ b2,
    const float* __restrict__ f0p, const float* __restrict__ a0p, const float* __restrict__ d0p,
    const float* __restrict__ f1p, const float* __restrict__ a1p, const float* __restrict__ d1p,
    const float* __restrict__ f2p, const float* __restrict__ a2p, const float* __restrict__ d2p,
    float* __restrict__ out)
{
  extern __shared__ _Float16 lds[];
  _Float16* xhi  = lds;              // 4 mt * 8 kt * 512 = 16384 halfs (32KB)
  _Float16* xlo  = xhi  + 16384;     // 32KB
  _Float16* h0hi = xlo  + 16384;     // 4 mt * 16 kt * 512 = 32768 halfs (64KB)
  _Float16* h1hi = h0hi + 32768;     // 64KB

  const int tid  = threadIdx.x;
  const int lane = tid & 31;
  const int wave = tid >> 5;
  const int row0 = blockIdx.x * MT;

  // ---- stage x tile -> LDS hi/lo, A-fragment swizzled (float4 granular) ----
  for (int i = tid; i < MT * IN_DIM / 4; i += TPB) {
    int m = i >> 6;              // 64 float4 per 256-wide row
    int k = (i & 63) << 2;       // 4-aligned: 4 elems share lane, consecutive j
    float4 v = *reinterpret_cast<const float4*>(x + (size_t)(row0 + m) * IN_DIM + k);
    int off = a_off(m, k, 8);
    float e[4] = {v.x, v.y, v.z, v.w};
    v4h hi, lo;
    #pragma unroll
    for (int j = 0; j < 4; ++j) {
      _Float16 h = (_Float16)e[j];
      hi[j] = h;
      lo[j] = (_Float16)(e[j] - (float)h);
    }
    *reinterpret_cast<v4h*>(xhi + off) = hi;
    *reinterpret_cast<v4h*>(xlo + off) = lo;
  }
  __syncthreads();

  const float f0 = *f0p, A0 = *a0p, D0 = *d0p;
  const float f1 = *f1p, A1 = *a1p, D1 = *d1p;
  const float f2 = *f2p, A2 = *a2p, D2 = *d2p;

  // ---------------- layer 0: h0 = act(x @ W0m + b0) ----------------
  for (int t = 0; t < 4; ++t) {
    const int ntile = wave * 4 + t;
    v8f acc[4] = {{0.f,0.f,0.f,0.f,0.f,0.f,0.f,0.f}, {0.f,0.f,0.f,0.f,0.f,0.f,0.f,0.f},
                  {0.f,0.f,0.f,0.f,0.f,0.f,0.f,0.f}, {0.f,0.f,0.f,0.f,0.f,0.f,0.f,0.f}};
    gemm_4m(acc, xhi, xlo, 8, ws + W0HI, ws + W0LO, ntile, 32, lane, true);
    const int n = (ntile << 4) + (lane & 15);
    const float bias = b0[n];
    #pragma unroll
    for (int m = 0; m < 4; ++m) {
      #pragma unroll
      for (int r = 0; r < 8; ++r) {
        int   mloc = (m << 4) + r + ((lane >> 4) << 3);
        float hv   = sine_act(acc[m][r] + bias, f0, A0, D0);
        h0hi[a_off(mloc, n, 16)] = (_Float16)hv;
        out[(size_t)(row0 + mloc) * (3 * UNITS) + n] = hv;
      }
    }
  }
  __syncthreads();

  // -------- layer 1: h1 = act(h0 @ W1m + b1) + x @ S1m --------
  for (int t = 0; t < 4; ++t) {
    const int ntile = wave * 4 + t;
    v8f skip[4] = {{0.f,0.f,0.f,0.f,0.f,0.f,0.f,0.f}, {0.f,0.f,0.f,0.f,0.f,0.f,0.f,0.f},
                   {0.f,0.f,0.f,0.f,0.f,0.f,0.f,0.f}, {0.f,0.f,0.f,0.f,0.f,0.f,0.f,0.f}};
    gemm_4m(skip, xhi, xlo, 8, ws + S1HI, ws + S1LO, ntile, 32, lane, true);
    v8f acc[4] = {{0.f,0.f,0.f,0.f,0.f,0.f,0.f,0.f}, {0.f,0.f,0.f,0.f,0.f,0.f,0.f,0.f},
                  {0.f,0.f,0.f,0.f,0.f,0.f,0.f,0.f}, {0.f,0.f,0.f,0.f,0.f,0.f,0.f,0.f}};
    gemm_4m(acc, h0hi, h0hi, 16, ws + W1HI, ws + W1LO, ntile, 32, lane, false);
    const int n = (ntile << 4) + (lane & 15);
    const float bias = b1[n];
    #pragma unroll
    for (int m = 0; m < 4; ++m) {
      #pragma unroll
      for (int r = 0; r < 8; ++r) {
        int   mloc = (m << 4) + r + ((lane >> 4) << 3);
        float hv   = sine_act(acc[m][r] + bias, f1, A1, D1) + skip[m][r];
        h1hi[a_off(mloc, n, 16)] = (_Float16)hv;
        out[(size_t)(row0 + mloc) * (3 * UNITS) + UNITS + n] = hv;
      }
    }
  }
  __syncthreads();

  // -------- layer 2: h2 = act(h1 @ W2m + b2) + x @ S2m --------
  for (int t = 0; t < 4; ++t) {
    const int ntile = wave * 4 + t;
    v8f skip[4] = {{0.f,0.f,0.f,0.f,0.f,0.f,0.f,0.f}, {0.f,0.f,0.f,0.f,0.f,0.f,0.f,0.f},
                   {0.f,0.f,0.f,0.f,0.f,0.f,0.f,0.f}, {0.f,0.f,0.f,0.f,0.f,0.f,0.f,0.f}};
    gemm_4m(skip, xhi, xlo, 8, ws + S2HI, ws + S2LO, ntile, 32, lane, true);
    v8f acc[4] = {{0.f,0.f,0.f,0.f,0.f,0.f,0.f,0.f}, {0.f,0.f,0.f,0.f,0.f,0.f,0.f,0.f},
                  {0.f,0.f,0.f,0.f,0.f,0.f,0.f,0.f}, {0.f,0.f,0.f,0.f,0.f,0.f,0.f,0.f}};
    gemm_4m(acc, h1hi, h1hi, 16, ws + W2HI, ws + W2LO, ntile, 32, lane, false);
    const int n = (ntile << 4) + (lane & 15);
    const float bias = b2[n];
    #pragma unroll
    for (int m = 0; m < 4; ++m) {
      #pragma unroll
      for (int r = 0; r < 8; ++r) {
        int   mloc = (m << 4) + r + ((lane >> 4) << 3);
        float hv   = sine_act(acc[m][r] + bias, f2, A2, D2) + skip[m][r];
        out[(size_t)(row0 + mloc) * (3 * UNITS) + 2 * UNITS + n] = hv;
      }
    }
  }
}

extern "C" void kernel_launch(void* const* d_in, const int* in_sizes, int n_in,
                              void* d_out, int out_size, void* d_ws, size_t ws_size,
                              hipStream_t stream) {
  const float* x  = (const float*)d_in[0];
  const float* W0 = (const float*)d_in[1];
  const float* b0 = (const float*)d_in[2];
  const float* M0 = (const float*)d_in[3];
  const float* f0 = (const float*)d_in[4];
  const float* a0 = (const float*)d_in[5];
  const float* d0 = (const float*)d_in[6];
  const float* W1 = (const float*)d_in[7];
  const float* b1 = (const float*)d_in[8];
  const float* M1 = (const float*)d_in[9];
  const float* f1 = (const float*)d_in[10];
  const float* a1 = (const float*)d_in[11];
  const float* d1 = (const float*)d_in[12];
  const float* S1 = (const float*)d_in[13];
  const float* SM1= (const float*)d_in[14];
  const float* W2 = (const float*)d_in[15];
  const float* b2 = (const float*)d_in[16];
  const float* M2 = (const float*)d_in[17];
  const float* f2 = (const float*)d_in[18];
  const float* a2 = (const float*)d_in[19];
  const float* d2 = (const float*)d_in[20];
  const float* S2 = (const float*)d_in[21];
  const float* SM2= (const float*)d_in[22];
  float* out = (float*)d_out;
  _Float16* ws = (_Float16*)d_ws;   // needs 3.5 MB

  // Mask + split + swizzle weights into L2-resident fragment planes.
  const int KN_s = IN_DIM * UNITS;  // 131072
  const int KN_u = UNITS * UNITS;   // 262144
  pack_weights<<<(KN_s + 255) / 256, 256, 0, stream>>>(W0, M0, ws + W0HI, ws + W0LO, IN_DIM, UNITS);
  pack_weights<<<(KN_u + 255) / 256, 256, 0, stream>>>(W1, M1, ws + W1HI, ws + W1LO, UNITS, UNITS);
  pack_weights<<<(KN_u + 255) / 256, 256, 0, stream>>>(W2, M2, ws + W2HI, ws + W2LO, UNITS, UNITS);
  pack_weights<<<(KN_s + 255) / 256, 256, 0, stream>>>(S1, SM1, ws + S1HI, ws + S1LO, IN_DIM, UNITS);
  pack_weights<<<(KN_s + 255) / 256, 256, 0, stream>>>(S2, SM2, ws + S2HI, ws + S2LO, IN_DIM, UNITS);

  const size_t smem = (size_t)(16384 * 2 + 32768 * 2) * sizeof(_Float16); // 192 KB
  reservoir_fused<<<dim3(BATCH / MT), TPB, smem, stream>>>(
      x, ws, b0, b1, b2, f0, a0, d0, f1, a1, d1, f2, a2, d2, out);
}
